// GAT_78572131713120
// MI455X (gfx1250) — matrix operand order repaired and checked by reading, not compiled, not run on hardware
//
#include <hip/hip_runtime.h>
#include <hip/hip_bf16.h>

typedef __attribute__((ext_vector_type(16))) _Float16 v16h;
typedef __attribute__((ext_vector_type(2)))  __fp16   v2fp;   // cvt_pkrtz return type
typedef __attribute__((ext_vector_type(8)))  float    v8f;

#define N_NODES  8192
#define EMB      128
#define HID      64
#define NHEADS   4
#define ALPHA    0.2f
#define MASK_FILL 9e-15f
#define LOG2E    1.4426950408889634f
#define JCHUNKS  (N_NODES / 32)   // 256
#define PF_DIST  8                // adj prefetch distance in 32-col chunks (1KB/row)

// ---------------------------------------------------------------------------
// Kernel 1: per (head,row) workgroup of 64 threads.
//   h = x @ W[head]         -> packed f16 in WMMA-B lane layout
//   f1 = h . a[:HID], f2 = h . a[HID:]   (both pre-scaled by log2(e))
// hB layout: [head][jc=row/32][ct=col/16][lane=col%16+16*(row/8%2)][s=row%8+8*(row/16%2)] halves,
// so the attention kernel loads one v16h (32B) per lane per B tile.
// ---------------------------------------------------------------------------
__global__ __launch_bounds__(64)
void gat_proj_kernel(const float* __restrict__ x,
                     const float* __restrict__ W,
                     const float* __restrict__ a,
                     _Float16* __restrict__ hB,
                     float* __restrict__ f1s,
                     float* __restrict__ f2s) {
    __shared__ float xs[EMB];
    __shared__ float p1[HID];
    __shared__ float p2[HID];

    const int head = blockIdx.x / N_NODES;
    const int row  = blockIdx.x % N_NODES;
    const int t    = threadIdx.x;            // 0..63 = output column

    for (int k = t; k < EMB; k += HID) xs[k] = x[(size_t)row * EMB + k];
    __syncthreads();

    const float* Wh = W + (size_t)head * EMB * HID;
    float acc = 0.f;
#pragma unroll 8
    for (int k = 0; k < EMB; ++k)
        acc = fmaf(xs[k], Wh[(size_t)k * HID + t], acc);

    // pack into WMMA-B-ready layout
    const int jc = row >> 5, kin = row & 31, ct = t >> 4, n = t & 15;
    const size_t off = ((((size_t)head * JCHUNKS + jc) * 4 + ct) * 32 + (n + ((kin >> 3) & 1) * 16)) * 16 + ((kin & 7) + ((kin >> 4) << 3));
    hB[off] = (_Float16)acc;

    const float* ah = a + (size_t)head * 2 * HID;   // a[head][:,0]
    p1[t] = acc * ah[t];
    p2[t] = acc * ah[HID + t];
    __syncthreads();
    for (int s = HID / 2; s > 0; s >>= 1) {
        if (t < s) { p1[t] += p1[t + s]; p2[t] += p2[t + s]; }
        __syncthreads();
    }
    if (t == 0) {
        f1s[(size_t)head * N_NODES + row] = p1[0] * LOG2E;
        f2s[(size_t)head * N_NODES + row] = p2[0] * LOG2E;
    }
}

// ---------------------------------------------------------------------------
// Kernel 2: per-head max over f2 (for the softmax shift upper bound)
// ---------------------------------------------------------------------------
__global__ __launch_bounds__(256)
void gat_f2max_kernel(const float* __restrict__ f2s, float* __restrict__ M2) {
    __shared__ float red[256];
    const int head = blockIdx.x;
    float m = -1e30f;
    for (int j = threadIdx.x; j < N_NODES; j += 256)
        m = fmaxf(m, f2s[(size_t)head * N_NODES + j]);
    red[threadIdx.x] = m;
    __syncthreads();
    for (int s = 128; s > 0; s >>= 1) {
        if (threadIdx.x < s) red[threadIdx.x] = fmaxf(red[threadIdx.x], red[threadIdx.x + s]);
        __syncthreads();
    }
    if (threadIdx.x == 0) M2[head] = red[0];
}

// ---------------------------------------------------------------------------
// Kernel 3: fused streaming attention. One block = 16 output rows, 4 waves,
// wave w = head w. Single pass over adj; exp-weights built per-lane in the
// exact 16x32 f16 A-fragment layout; 5x v_wmma_f32_16x16x32_f16 per chunk:
// 4 against h-tiles (numerator) + 1 against an all-ones tile (row-sum
// denominator on the matrix pipe -> no VALU add tree, no epilogue shuffles,
// and C4 shares the C-layout so inv[v] lines up with C0..C3[v] per lane).
// All chunk inputs (adj, f2, B tiles) are software-pipelined one chunk ahead.
// ---------------------------------------------------------------------------
__global__ __launch_bounds__(128)
void gat_attn_kernel(const int* __restrict__ adj,
                     const _Float16* __restrict__ hB,
                     const float* __restrict__ f1s,
                     const float* __restrict__ f2s,
                     const float* __restrict__ M2,
                     float* __restrict__ out) {
    const int tid = threadIdx.x;
    const int w   = tid >> 5;            // head (wave id)
    const int L   = tid & 31;            // lane
    const int i0  = blockIdx.x * 16;
    const int M   = L & 15;              // A-matrix row this lane owns
    const int klo = (L < 16) ? 0 : 8;    // A-layout K-octet base
    const int row = i0 + M;

    // softmax shift constant: c >= rowmax, softmax is shift-invariant.
    const float f1v = f1s[(size_t)w * N_NODES + row];   // already * log2e
    const float m2  = M2[w];
    const float MFs = MASK_FILL * LOG2E;
    const float tub = f1v + m2;
    const float c   = fmaxf(MFs, fmaxf(tub, tub * ALPHA));
    const float mfe = __builtin_amdgcn_exp2f(MFs - c);  // masked-entry weight
    const float f1c = f1v - c;                          // folded shift
    const float cc  = c * (ALPHA - 1.0f);               // LRelu(t)-c = max(t-c, a*(t-c)+cc)

    const int*   adjrow = adj + (size_t)row * N_NODES;
    const float* f2h    = f2s + (size_t)w * N_NODES;
    const v16h*  hBw    = (const v16h*)hB + (size_t)w * JCHUNKS * 4 * 32;

    // all-ones B tile for the denominator WMMA
    v16h onesB;
#pragma unroll
    for (int p = 0; p < 16; ++p) onesB[p] = (_Float16)1.0f;

    v8f C0 = {}, C1 = {}, C2 = {}, C3 = {}, C4 = {};

    // ---- pipeline prologue: load chunk 0 ----
    int4   cA0 = *(const int4*)(adjrow + klo);
    int4   cA1 = *(const int4*)(adjrow + klo + 4);
    int4   cA2 = *(const int4*)(adjrow + klo + 16);
    int4   cA3 = *(const int4*)(adjrow + klo + 20);
    float4 cF0 = *(const float4*)(f2h + klo);
    float4 cF1 = *(const float4*)(f2h + klo + 4);
    float4 cF2 = *(const float4*)(f2h + klo + 16);
    float4 cF3 = *(const float4*)(f2h + klo + 20);
    v16h B0 = hBw[L], B1 = hBw[32 + L], B2 = hBw[64 + L], B3 = hBw[96 + L];

#pragma unroll 2
    for (int jc = 0; jc < JCHUNKS; ++jc) {
        // ---- issue next-chunk loads first (clamped; uniform, branch-free) ----
        const int jn  = (jc + 1 < JCHUNKS) ? jc + 1 : jc;
        const int j0n = jn * 32;
        const int4   nA0 = *(const int4*)(adjrow + j0n + klo);
        const int4   nA1 = *(const int4*)(adjrow + j0n + klo + 4);
        const int4   nA2 = *(const int4*)(adjrow + j0n + klo + 16);
        const int4   nA3 = *(const int4*)(adjrow + j0n + klo + 20);
        const float4 nF0 = *(const float4*)(f2h + j0n + klo);
        const float4 nF1 = *(const float4*)(f2h + j0n + klo + 4);
        const float4 nF2 = *(const float4*)(f2h + j0n + klo + 16);
        const float4 nF3 = *(const float4*)(f2h + j0n + klo + 20);
        const v16h* bpn = hBw + (size_t)jn * 4 * 32 + L;
        const v16h nB0 = bpn[0];
        const v16h nB1 = bpn[32];
        const v16h nB2 = bpn[64];
        const v16h nB3 = bpn[96];

        // branch-free clamped look-ahead prefetch of the adj HBM stream
        // (locality 3 -> WGP-scope: pull into all cache levels)
        const int jp = (jc + PF_DIST < JCHUNKS) ? jc + PF_DIST : JCHUNKS - 1;
        __builtin_prefetch(adjrow + jp * 32 + klo, 0, 3); // global_prefetch_b8

        // ---- build A fragment from current chunk (hides next-load latency) ----
        const int   am[16] = {cA0.x, cA0.y, cA0.z, cA0.w, cA1.x, cA1.y, cA1.z, cA1.w,
                              cA2.x, cA2.y, cA2.z, cA2.w, cA3.x, cA3.y, cA3.z, cA3.w};
        const float fv[16] = {cF0.x, cF0.y, cF0.z, cF0.w, cF1.x, cF1.y, cF1.z, cF1.w,
                              cF2.x, cF2.y, cF2.z, cF2.w, cF3.x, cF3.y, cF3.z, cF3.w};
        float wv[16];
#pragma unroll
        for (int k = 0; k < 16; ++k) {
            const float tc = f1c + fv[k];                      // (f1+f2)*log2e - c
            const float l  = fmaxf(tc, fmaf(tc, ALPHA, cc));   // LRelu(t)-c, folded
            const float e  = __builtin_amdgcn_exp2f(l);
            wv[k] = (am[k] > 0) ? e : mfe;
        }

        union { v16h v; v2fp h2[8]; } A;
#pragma unroll
        for (int p = 0; p < 8; ++p)
            A.h2[p] = __builtin_amdgcn_cvt_pkrtz(wv[2 * p], wv[2 * p + 1]);

        // ---- 5 WMMAs: 4 numerator tiles + row-sum denominator ----
        C0 = __builtin_amdgcn_wmma_f32_16x16x32_f16(false, A.v, false, B0, (short)0, C0, false, false);
        C1 = __builtin_amdgcn_wmma_f32_16x16x32_f16(false, A.v, false, B1, (short)0, C1, false, false);
        C2 = __builtin_amdgcn_wmma_f32_16x16x32_f16(false, A.v, false, B2, (short)0, C2, false, false);
        C3 = __builtin_amdgcn_wmma_f32_16x16x32_f16(false, A.v, false, B3, (short)0, C3, false, false);
        C4 = __builtin_amdgcn_wmma_f32_16x16x32_f16(false, A.v, false, onesB, (short)0, C4, false, false);

        // ---- rotate pipeline registers ----
        cA0 = nA0; cA1 = nA1; cA2 = nA2; cA3 = nA3;
        cF0 = nF0; cF1 = nF1; cF2 = nF2; cF3 = nF3;
        B0 = nB0; B1 = nB1; B2 = nB2; B3 = nB3;
    }

    // C layout: lane<16 -> rows 0..7, lane>=16 -> rows 8..15; col = L%16.
    // C4[v] is the softmax denominator for exactly the row C0..C3[v] belong to.
    const int mbase = (L < 16) ? 0 : 8;
    const int ncol  = L & 15;
#pragma unroll
    for (int v = 0; v < 8; ++v) {
        const float inv  = __builtin_amdgcn_rcpf(C4[v]);
        const int   orow = i0 + mbase + v;
        float* op = out + (size_t)orow * (NHEADS * HID) + w * HID + ncol;
        float y;
        y = C0[v] * inv; op[0]  = (y > 0.f) ? y : (__builtin_amdgcn_exp2f(y * LOG2E) - 1.f);
        y = C1[v] * inv; op[16] = (y > 0.f) ? y : (__builtin_amdgcn_exp2f(y * LOG2E) - 1.f);
        y = C2[v] * inv; op[32] = (y > 0.f) ? y : (__builtin_amdgcn_exp2f(y * LOG2E) - 1.f);
        y = C3[v] * inv; op[48] = (y > 0.f) ? y : (__builtin_amdgcn_exp2f(y * LOG2E) - 1.f);
    }
}

// ---------------------------------------------------------------------------
extern "C" void kernel_launch(void* const* d_in, const int* in_sizes, int n_in,
                              void* d_out, int out_size, void* d_ws, size_t ws_size,
                              hipStream_t stream) {
    const float* x   = (const float*)d_in[0];
    const int*   adj = (const int*)d_in[1];
    const float* W   = (const float*)d_in[2];
    const float* a   = (const float*)d_in[3];
    float*       out = (float*)d_out;

    char* ws = (char*)d_ws;
    _Float16* hB  = (_Float16*)ws;                                   // 4 MB
    float*    f1s = (float*)(ws + (size_t)NHEADS * N_NODES * HID * 2);
    float*    f2s = f1s + (size_t)NHEADS * N_NODES;
    float*    M2  = f2s + (size_t)NHEADS * N_NODES;

    gat_proj_kernel<<<NHEADS * N_NODES, 64, 0, stream>>>(x, W, a, hB, f1s, f2s);
    gat_f2max_kernel<<<NHEADS, 256, 0, stream>>>(f2s, M2);
    gat_attn_kernel<<<N_NODES / 16, 128, 0, stream>>>(adj, hB, f1s, f2s, M2, out);
}
